// NonLocalBlock_71588514890083
// MI455X (gfx1250) — matrix operand order repaired and checked by reading, not compiled, not run on hardware
//
#include <hip/hip_runtime.h>
#include <hip/hip_bf16.h>

// ---------------- problem constants ----------------
#define BB  4
#define CC  256
#define CII 128
#define NN  4096          // H*W = 64*64
#define EPSV 1e-5f

typedef __attribute__((ext_vector_type(16))) _Float16 v16h;
typedef __attribute__((ext_vector_type(8)))  _Float16 v8h;
typedef __attribute__((ext_vector_type(8)))  float    v8f;

#if __has_builtin(__builtin_amdgcn_global_load_async_to_lds_b128) && \
    __has_builtin(__builtin_amdgcn_s_wait_asynccnt)
#define USE_ASYNC_LDS 1
typedef int i4v __attribute__((__vector_size__(4 * sizeof(int))));
typedef __attribute__((address_space(1))) i4v gi4v;   // global (AS1) int4
typedef __attribute__((address_space(3))) i4v li4v;   // LDS    (AS3) int4
#define WAIT_ASYNC(n) __builtin_amdgcn_s_wait_asynccnt(n)
#else
#define USE_ASYNC_LDS 0
#define WAIT_ASYNC(n)
#endif

// async (or fallback sync) 16-byte global -> LDS copy
__device__ __forceinline__ void copy16_g2l(const _Float16* __restrict__ g,
                                           _Float16* l) {
#if USE_ASYNC_LDS
    __builtin_amdgcn_global_load_async_to_lds_b128((gi4v*)g, (li4v*)l, 0, 0);
#else
    *(v8h*)l = *(const v8h*)g;
#endif
}

// =====================================================================
// Fragment loaders (wave32); k order is contiguous, not the ISA 7.12.2 table order.
// A (16x32 f16, MxK): lane m = lane&15 holds row M=m; K = half*16 + e,
//   e = 0..15 (contiguous, matching the B fragments).
// =====================================================================
__device__ __forceinline__ v16h load_a_16x32(const _Float16* __restrict__ base,
                                             int ld, int row0, int k0) {
    const int lane = threadIdx.x & 31;
    const _Float16* p = base + (size_t)(row0 + (lane & 15)) * ld + k0 + ((lane >> 4) << 4);
    v8h g0 = *(const v8h*)(p);        // K = half*16 .. half*16+7
    v8h g1 = *(const v8h*)(p + 8);    // K = half*16+8 .. half*16+15
    v16h a;
#pragma unroll
    for (int i = 0; i < 8; ++i) { a[i] = g0[i]; a[8 + i] = g1[i]; }
    return a;
}

// B (32x16 f16, KxN): lane holds column n = lane&15; rows K = (lane>>4)*16 + e,
// e = 0..15 packed two-per-VGPR.  Element = base[(n0+n)*ld + k0 + K].
__device__ __forceinline__ v16h load_b_32x16(const _Float16* __restrict__ base,
                                             int ld, int n0, int k0) {
    const int lane = threadIdx.x & 31;
    const _Float16* p = base + (size_t)(n0 + (lane & 15)) * ld + k0 + ((lane >> 4) << 4);
    v8h g0 = *(const v8h*)(p);
    v8h g1 = *(const v8h*)(p + 8);
    v16h b;
#pragma unroll
    for (int i = 0; i < 8; ++i) { b[i] = g0[i]; b[8 + i] = g1[i]; }
    return b;
}

__device__ __forceinline__ v8f wmma_f16(v16h a, v16h b, v8f c) {
    return __builtin_amdgcn_wmma_f32_16x16x32_f16(false, a, false, b, (short)0, c,
                                                  false, false);
}

// =====================================================================
// K0: convert W_w (C x Ci, f32) -> f16 copy
// =====================================================================
__global__ void convert_w_kernel(const float* __restrict__ W_w,
                                 _Float16* __restrict__ wh) {
    int i = blockIdx.x * 256 + threadIdx.x;
    if (i < CC * CII) wh[i] = (_Float16)W_w[i];
}

// =====================================================================
// K1: stage x[:, n0:n0+32] in LDS; emit xh[B][N][C] f16 (Q rows / K rows)
//     and g-projection gxT[B][Ci][N] f16 (contiguous along key => PV B-frags)
// =====================================================================
__global__ __launch_bounds__(256) void prep_kernel(const float* __restrict__ x,
                                                   const float* __restrict__ g_w,
                                                   const float* __restrict__ g_b,
                                                   _Float16* __restrict__ xh,
                                                   _Float16* __restrict__ gxT) {
    __shared__ float xt[CC][33];
    const int t = threadIdx.x;
    const int blk = blockIdx.x;                 // B * N/32 = 512 blocks
    const int b  = blk / (NN / 32);
    const int n0 = (blk % (NN / 32)) * 32;
    const float* xb = x + (size_t)b * CC * NN;

    for (int idx = t; idx < CC * 32; idx += 256) {
        int c = idx >> 5, j = idx & 31;
        xt[c][j] = xb[(size_t)c * NN + n0 + j];
    }
    __syncthreads();

    for (int idx = t; idx < 32 * CC; idx += 256) {
        int c = idx & (CC - 1), j = idx >> 8;
        xh[((size_t)b * NN + n0 + j) * CC + c] = (_Float16)xt[c][j];
    }

    const int j = t & 31, g = t >> 5;           // wave g computes ci in [g*16, g*16+16)
    float acc[16];
#pragma unroll
    for (int i = 0; i < 16; ++i) acc[i] = g_b[g * 16 + i];
    for (int c = 0; c < CC; ++c) {
        float xv = xt[c][j];
#pragma unroll
        for (int i = 0; i < 16; ++i)
            acc[i] += g_w[(g * 16 + i) * CC + c] * xv;   // wave-uniform -> s_load
    }
#pragma unroll
    for (int i = 0; i < 16; ++i)
        gxT[((size_t)b * CII + g * 16 + i) * NN + n0 + j] = (_Float16)acc[i];
}

// =====================================================================
// K2: fused flash-attention with double-buffered async global->LDS
// staging of the K/V tiles (shared by the 8 waves of the block).
// =====================================================================
#define KROW_LD 264     // 256 + 8 pad halfs  (row = one key, 512B payload)
#define VROW_LD 40      // 32 + 8 pad halfs   (row = one ci channel)
#define NT      (NN / 32)

__global__ __launch_bounds__(256) void attn_kernel(const _Float16* __restrict__ xh,
                                                   const _Float16* __restrict__ gxT,
                                                   _Float16* __restrict__ y) {
    const int wave = threadIdx.x >> 5;
    const int lane = threadIdx.x & 31;
    const int t    = threadIdx.x;
    const int wid  = blockIdx.x * 8 + wave;          // 1024 wave-tiles
    const int b    = wid / (NN / 16);
    const int q0   = (wid % (NN / 16)) * 16;

    const _Float16* xq = xh  + (size_t)b * NN * CC;  // Q rows / K rows
    const _Float16* vt = gxT + (size_t)b * CII * NN; // V^T (Ci x N)

    __shared__ __align__(16) _Float16 kbuf[2][32][KROW_LD];   // 33 KB
    __shared__ __align__(16) _Float16 vbuf[2][CII][VROW_LD];  // 20 KB
    __shared__ __align__(16) _Float16 pbuf[8][16][32];        //  8 KB

    // ---- cooperative async prefetch of one 32-key tile (6 loads/thread) ----
    auto prefetch = [&](int buf, int key0) {
#pragma unroll
        for (int i = 0; i < 4; ++i) {                // K tile: 32 x 512B
            int e = t + i * 256;                     // 1024 x 16B segments
            int row = e >> 5, seg = e & 31;
            copy16_g2l(xq + (size_t)(key0 + row) * CC + seg * 8,
                       &kbuf[buf][row][seg * 8]);
        }
#pragma unroll
        for (int i = 0; i < 2; ++i) {                // V tile: 128 x 64B
            int e = t + i * 256;                     // 512 x 16B segments
            int row = e >> 2, seg = e & 3;
            copy16_g2l(vt + (size_t)row * NN + key0 + seg * 8,
                       &vbuf[buf][row][seg * 8]);
        }
    };

    prefetch(0, 0);

    // Q fragments resident: 16 x 256 f16 = 64 VGPRs
    v16h qf[8];
#pragma unroll
    for (int kk = 0; kk < 8; ++kk) qf[kk] = load_a_16x32(xq, CC, q0, kk * 32);

    v8f o[8] = {};
    float m[8], l[8];
#pragma unroll
    for (int r = 0; r < 8; ++r) { m[r] = -1e30f; l[r] = 0.f; }

    for (int it = 0; it < NT; ++it) {
        const int key0 = it * 32;
        const int cur = it & 1;
        __syncthreads();                       // buf[cur^1] readers (iter it-1) done
        if (it + 1 < NT) {
            prefetch(cur ^ 1, key0 + 32);
            WAIT_ASYNC(6);                     // tile `it` landed; 6 in flight for it+1
        } else {
            WAIT_ASYNC(0);
        }
        __syncthreads();                       // all waves' tile-`it` data visible

        // ---- S tile = Q (16x256) . K^T (256x32) from LDS ----
        v8f s0 = {}, s1 = {};
        const _Float16* kb = &kbuf[cur][0][0];
#pragma unroll
        for (int kk = 0; kk < 8; ++kk) {
            v16h b0 = load_b_32x16(kb, KROW_LD, 0,  kk * 32);
            v16h b1 = load_b_32x16(kb, KROW_LD, 16, kk * 32);
            s0 = wmma_f16(qf[kk], b0, s0);
            s1 = wmma_f16(qf[kk], b1, s1);
        }

        // ---- online softmax per row (row = r + 8*(lane>=16)) ----
#pragma unroll
        for (int r = 0; r < 8; ++r) {
            float rm = fmaxf(s0[r], s1[r]);
#pragma unroll
            for (int off = 8; off >= 1; off >>= 1)
                rm = fmaxf(rm, __shfl_xor(rm, off, 16));
            float mn    = fmaxf(m[r], rm);
            float scale = __expf(m[r] - mn);
            float p0    = __expf(s0[r] - mn);
            float p1    = __expf(s1[r] - mn);
            float rs    = p0 + p1;
#pragma unroll
            for (int off = 8; off >= 1; off >>= 1)
                rs += __shfl_xor(rs, off, 16);
            l[r] = l[r] * scale + rs;
            m[r] = mn;
#pragma unroll
            for (int jj = 0; jj < 8; ++jj) o[jj][r] *= scale;

            int row = r + ((lane >> 4) << 3);  // C-layout -> row-major staging
            pbuf[wave][row][lane & 15]        = (_Float16)p0;
            pbuf[wave][row][16 + (lane & 15)] = (_Float16)p1;
        }
#if __has_builtin(__builtin_amdgcn_s_wait_dscnt)
        __builtin_amdgcn_s_wait_dscnt(0);      // same-wave DS store->load
#else
        __builtin_amdgcn_fence(__ATOMIC_ACQ_REL, "wavefront");
#endif
        // ---- O += P (16x32) . V (32x128), V from LDS (ci-major) ----
        v16h pa = load_a_16x32(&pbuf[wave][0][0], 32, 0, 0);
#pragma unroll
        for (int jj = 0; jj < 8; ++jj) {
            // B[k][n]: n = ci = jj*16 + (lane&15), K contiguous along key
            const _Float16* p = &vbuf[cur][jj * 16 + (lane & 15)][(lane >> 4) << 4];
            v8h g0 = *(const v8h*)(p);
            v8h g1 = *(const v8h*)(p + 8);
            v16h vb;
#pragma unroll
            for (int i = 0; i < 8; ++i) { vb[i] = g0[i]; vb[8 + i] = g1[i]; }
            o[jj] = wmma_f16(pa, vb, o[jj]);
        }
    }

    // ---- normalize, store y[B][N][Ci] f16 ----
#pragma unroll
    for (int r = 0; r < 8; ++r) {
        float inv = 1.f / l[r];
        int q = q0 + r + ((lane >> 4) << 3);
#pragma unroll
        for (int jj = 0; jj < 8; ++jj)
            y[((size_t)b * NN + q) * CII + jj * 16 + (lane & 15)] =
                (_Float16)(o[jj][r] * inv);
    }
}

// =====================================================================
// Shared tile: wy[p0:p0+16, o0:o0+16] = y . W_w^T + W_b  (WMMA, K=128)
// =====================================================================
__device__ __forceinline__ v8f compute_wy_tile(const _Float16* __restrict__ y,
                                               const _Float16* __restrict__ wh,
                                               const float* __restrict__ W_b,
                                               int p0, int o0) {
    v8f acc = {};
#pragma unroll
    for (int kk = 0; kk < 4; ++kk) {
        v16h a = load_a_16x32(y,  CII, p0, kk * 32);
        v16h b = load_b_32x16(wh, CII, o0, kk * 32);
        acc = wmma_f16(a, b, acc);
    }
    float bias = W_b[o0 + (threadIdx.x & 15)];
#pragma unroll
    for (int r = 0; r < 8; ++r) acc[r] += bias;
    return acc;
}

// =====================================================================
// K3: per-channel sum / sumsq of wy (wy recomputed, never stored)
// =====================================================================
__global__ __launch_bounds__(256) void stats_kernel(const _Float16* __restrict__ y,
                                                    const _Float16* __restrict__ wh,
                                                    const float* __restrict__ W_b,
                                                    float* __restrict__ sums,
                                                    float* __restrict__ sqs) {
    __shared__ float ls[CC], lq[CC];
    const int t = threadIdx.x, wave = t >> 5, lane = t & 31;
    const int p0 = blockIdx.x * 16;
#pragma unroll
    for (int half = 0; half < 2; ++half) {
        int o0 = wave * 32 + half * 16;              // 8 waves x 2 = all 256 chans
        v8f acc = compute_wy_tile(y, wh, W_b, p0, o0);
        float s = 0.f, q = 0.f;
#pragma unroll
        for (int r = 0; r < 8; ++r) { s += acc[r]; q += acc[r] * acc[r]; }
        s += __shfl_xor(s, 16, 32);
        q += __shfl_xor(q, 16, 32);
        if (lane < 16) { ls[o0 + lane] = s; lq[o0 + lane] = q; }
    }
    __syncthreads();
    atomicAdd(&sums[t], ls[t]);
    atomicAdd(&sqs[t],  lq[t]);
}

// =====================================================================
// K4: recompute wy tile, BatchNorm (batch stats) + gamma/beta + residual
// =====================================================================
__global__ __launch_bounds__(256) void final_kernel(const _Float16* __restrict__ y,
                                                    const _Float16* __restrict__ wh,
                                                    const float* __restrict__ W_b,
                                                    const float* __restrict__ sums,
                                                    const float* __restrict__ sqs,
                                                    const float* __restrict__ gamma,
                                                    const float* __restrict__ beta,
                                                    const float* __restrict__ x,
                                                    float* __restrict__ out) {
    const int t = threadIdx.x, wave = t >> 5, lane = t & 31;
    const int p0 = blockIdx.x * 16;
    const float inv_bn = 1.f / (float)(BB * NN);
#pragma unroll
    for (int half = 0; half < 2; ++half) {
        int o0 = wave * 32 + half * 16;
        v8f acc = compute_wy_tile(y, wh, W_b, p0, o0);
        int ch = o0 + (lane & 15);
        float mean = sums[ch] * inv_bn;
        float var  = sqs[ch] * inv_bn - mean * mean;
        float rstd = rsqrtf(var + EPSV);
        float ga = gamma[ch], be = beta[ch];
        int b  = p0 / NN;
        int n0 = p0 % NN;
#pragma unroll
        for (int r = 0; r < 8; ++r) {
            int n = n0 + r + ((lane >> 4) << 3);
            size_t idx = ((size_t)(b * CC + ch)) * NN + n;
            out[idx] = (acc[r] - mean) * rstd * ga + be + x[idx];
        }
    }
}

// =====================================================================
extern "C" void kernel_launch(void* const* d_in, const int* in_sizes, int n_in,
                              void* d_out, int out_size, void* d_ws, size_t ws_size,
                              hipStream_t stream) {
    const float* x     = (const float*)d_in[0];
    const float* g_w   = (const float*)d_in[1];
    const float* g_b   = (const float*)d_in[2];
    const float* W_w   = (const float*)d_in[3];
    const float* W_b   = (const float*)d_in[4];
    const float* gamma = (const float*)d_in[5];
    const float* beta  = (const float*)d_in[6];
    float* out = (float*)d_out;

    // workspace layout (~16.8 MB)
    char* ws = (char*)d_ws;
    _Float16* xh  = (_Float16*)ws;                                   // 8 MiB
    _Float16* gxT = (_Float16*)(ws + (size_t)BB * NN * CC * 2);      // 4 MiB
    _Float16* y   = (_Float16*)(ws + (size_t)BB * NN * CC * 2
                                   + (size_t)BB * CII * NN * 2);     // 4 MiB
    _Float16* wh  = (_Float16*)((char*)y + (size_t)BB * NN * CII * 2);
    float* sums   = (float*)((char*)wh + (size_t)CC * CII * 2);
    float* sqs    = sums + CC;

    convert_w_kernel<<<(CC * CII + 255) / 256, 256, 0, stream>>>(W_w, wh);
    prep_kernel<<<BB * (NN / 32), 256, 0, stream>>>(x, g_w, g_b, xh, gxT);
    attn_kernel<<<(BB * NN / 16) / 8, 256, 0, stream>>>(xh, gxT, y);
    (void)hipMemsetAsync(sums, 0, 2 * CC * sizeof(float), stream);
    stats_kernel<<<(BB * NN) / 16, 256, 0, stream>>>(y, wh, W_b, sums, sqs);
    final_kernel<<<(BB * NN) / 16, 256, 0, stream>>>(y, wh, W_b, sums, sqs,
                                                     gamma, beta, x, out);
}
